// DeepSeekV3MoE_55061480735140
// MI455X (gfx1250) — compile-verified, matrix operand order checked by reading, not run on hardware
//
#include <hip/hip_runtime.h>
#include <cstdint>

typedef __bf16 bf16;
typedef __attribute__((ext_vector_type(16))) __bf16 v16bf;
typedef __attribute__((ext_vector_type(8)))  __bf16 v8bf;
typedef __attribute__((ext_vector_type(8)))  float  v8f;
typedef __attribute__((ext_vector_type(4)))  float  f32x4;

#define T_TOK 8192
#define HDIM  2048
#define IDIM  1408
#define NEXP  16
#define NV    17                      // 16 routed + 1 shared (virtual expert 16)
#define ROWS_TOTAL (3 * T_TOK)        // 2T routed slots + T shared slots

__device__ __forceinline__ float silu_f(float x) { return x / (1.0f + __expf(-x)); }

// Async-copy 64 contiguous bytes global -> LDS (4 x b128, ASYNCcnt-tracked).
// IOFFSET applies to both the LDS and global addresses, so consecutive 16B
// chunks land consecutively on both sides.
__device__ __forceinline__ void async_copy64B(uint32_t lds_off, const void* g) {
    asm volatile("global_load_async_to_lds_b128 %0, %1, off\n\t"
                 "global_load_async_to_lds_b128 %0, %1, off offset:16\n\t"
                 "global_load_async_to_lds_b128 %0, %1, off offset:32\n\t"
                 "global_load_async_to_lds_b128 %0, %1, off offset:48"
                 :: "v"(lds_off), "v"(g) : "memory");
}

__device__ __forceinline__ void wait_async0() {
    asm volatile("s_wait_asynccnt 0x0" ::: "memory");
}

// generic LDS pointer -> 32-bit LDS offset (flat LDS aperture keeps the
// wave-relative offset in addr[31:0])
__device__ __forceinline__ uint32_t lds_off_of(const void* p) {
    return (uint32_t)(uintptr_t)p;
}

// Load a 16x32 bf16 fragment (A/B layout, ISA 7.12.2) from a row-major LDS
// tile with row stride 64 bf16; kt selects the 32-wide k-slice (0 or 32).
__device__ __forceinline__ v16bf load_frag64(const bf16* tile, int rowBase, int kt, int lane) {
    const int r  = rowBase + (lane & 15);
    const int kh = (lane >> 4) << 3;            // 0 or 8
    const bf16* p = tile + r * 64 + kt + kh;
    v8bf lo = *(const v8bf*)(p);                // K = kh .. kh+7
    v8bf hi = *(const v8bf*)(p + 16);           // K = 16+kh .. 16+kh+7
    v16bf f;
#pragma unroll
    for (int i = 0; i < 8; ++i) { f[i] = lo[i]; f[i + 8] = hi[i]; }
    return f;
}

// ---------------- kernel 1: zero output + expert counters ----------------
__global__ void moe_zero_kernel(float* __restrict__ out, int* __restrict__ counts) {
    size_t i = (size_t)blockIdx.x * blockDim.x + threadIdx.x;
    size_t stride = (size_t)gridDim.x * blockDim.x;
    for (size_t j = i; j < (size_t)T_TOK * HDIM; j += stride) out[j] = 0.0f;
    if (i < NV) counts[i] = 0;
}

// ---------------- kernel 2: X fp32 -> bf16 ----------------
__global__ void moe_cvt_kernel(const float* __restrict__ x, bf16* __restrict__ xb) {
    size_t i = (size_t)blockIdx.x * blockDim.x + threadIdx.x;
    size_t stride = (size_t)gridDim.x * blockDim.x;
    for (size_t j = i; j < (size_t)T_TOK * HDIM; j += stride) xb[j] = (bf16)x[j];
}

// ---------------- kernel 3: router (one wave32 per token) ----------------
__global__ void moe_router_kernel(const float* __restrict__ x, const float* __restrict__ gw,
                                  const float* __restrict__ bias, int* __restrict__ topkIdx,
                                  float* __restrict__ topkGate, int* __restrict__ counts) {
    const int lane = threadIdx.x & 31;
    const int wv   = threadIdx.x >> 5;
    const int t    = blockIdx.x * 8 + wv;
    if (t >= T_TOK) return;
    const float* xr = x + (size_t)t * HDIM;
    float sc[NEXP];
#pragma unroll
    for (int e = 0; e < NEXP; ++e) {
        const float* wr = gw + (size_t)e * HDIM;
        float s = 0.0f;
        for (int h = lane; h < HDIM; h += 32) s += xr[h] * wr[h];
#pragma unroll
        for (int off = 16; off > 0; off >>= 1) s += __shfl_xor(s, off, 32);
        sc[e] = 1.0f / (1.0f + __expf(-s));     // sigmoid score (all lanes)
    }
    if (lane == 0) {
        int e0 = -1, e1 = -1; float r0 = -1e30f, r1 = -1e30f;
        for (int e = 0; e < NEXP; ++e) {
            float r = sc[e] + bias[e];
            if (r > r0)      { r1 = r0; e1 = e0; r0 = r; e0 = e; }
            else if (r > r1) { r1 = r;  e1 = e; }
        }
        float g0 = sc[e0], g1 = sc[e1];
        float inv = 1.0f / (g0 + g1);
        topkIdx[t * 2]      = e0;  topkIdx[t * 2 + 1] = e1;
        topkGate[t * 2]     = g0 * inv;
        topkGate[t * 2 + 1] = g1 * inv;
        atomicAdd(&counts[e0], 1);
        atomicAdd(&counts[e1], 1);
    }
}

// ---------------- kernel 4: prefix sum of counts -> bases ----------------
__global__ void moe_prefix_kernel(const int* __restrict__ counts, int* __restrict__ bases,
                                  int* __restrict__ cursors) {
    if (threadIdx.x == 0 && blockIdx.x == 0) {
        int acc = 0;
        for (int v = 0; v < NEXP; ++v) { bases[v] = acc; acc += counts[v]; }
        bases[NEXP] = acc;                       // shared-expert rows start here (== 2T)
        for (int v = 0; v < NV; ++v) cursors[v] = 0;
    }
}

// ---------------- kernel 5: scatter tokens into compact slot lists ----------------
__global__ void moe_scatter_kernel(const int* __restrict__ topkIdx, const float* __restrict__ topkGate,
                                   const int* __restrict__ bases, int* __restrict__ cursors,
                                   int* __restrict__ slotToken, float* __restrict__ slotGate) {
    int t = blockIdx.x * blockDim.x + threadIdx.x;
    if (t >= T_TOK) return;
#pragma unroll
    for (int k = 0; k < 2; ++k) {
        int e   = topkIdx[t * 2 + k];
        int pos = atomicAdd(&cursors[e], 1);
        int s   = bases[e] + pos;
        slotToken[s] = t;
        slotGate[s]  = topkGate[t * 2 + k];
    }
    int s = bases[NEXP] + t;                     // shared expert: identity list, gate 1
    slotToken[s] = t;
    slotGate[s]  = 1.0f;
}

// ---------------- kernel 6: grouped up-projection + SwiGLU ----------------
// Block tile: 128 slot-rows x 64 I-cols x KT=64, both W1 and W3 share the X tile.
// X tile arrives via global_load_async_to_lds_b128; weights convert f32->bf16
// through VGPRs. 16 x v_wmma_f32_16x16x32_bf16 per stage.
__global__ __launch_bounds__(256, 2)
void moe_up_kernel(const bf16* __restrict__ Xb,
                   const float* __restrict__ w1, const float* __restrict__ w3,
                   const float* __restrict__ sw1, const float* __restrict__ sw3,
                   const int* __restrict__ counts, const int* __restrict__ bases,
                   const int* __restrict__ slotToken, bf16* __restrict__ Act) {
    const int v     = blockIdx.z;
    const int n     = (v == NEXP) ? T_TOK : counts[v];
    const int rows0 = blockIdx.x * 128;
    if (rows0 >= n) return;                      // block-uniform: EXEC stays full
    const int n0      = blockIdx.y * 64;
    const int baseRow = bases[v];
    const float* w1p = (v == NEXP) ? sw1 : (w1 + (size_t)v * IDIM * HDIM);
    const float* w3p = (v == NEXP) ? sw3 : (w3 + (size_t)v * IDIM * HDIM);

    __shared__ __align__(32) bf16 smX[128 * 64];   // 16 KB
    __shared__ __align__(32) bf16 smW1[64 * 64];   //  8 KB
    __shared__ __align__(32) bf16 smW3[64 * 64];   //  8 KB
    __shared__ int smTok[128];

    const int tid = threadIdx.x;
    if (tid < 128) {
        int lr = rows0 + tid;
        int tok = (lr < n) ? slotToken[baseRow + lr] : 0;  // clamp: padded rows read row 0
        smTok[tid] = tok;
    }
    __syncthreads();

    const int lane = tid & 31;
    const int wv   = tid >> 5;
    const int wm   = (wv >> 1) * 32;             // 4 wave-rows
    const int wn   = (wv & 1) * 32;              // 2 wave-cols

    v8f accU[2][2], accV[2][2];
#pragma unroll
    for (int i = 0; i < 2; ++i)
#pragma unroll
        for (int j = 0; j < 2; ++j) { accU[i][j] = {}; accV[i][j] = {}; }

    const int xr = tid >> 1, xc = (tid & 1) * 32;   // X: 64B per thread (async)
    const int wr = tid >> 2, wc = (tid & 3) * 16;   // W: 16 f32->bf16 per thread

    const uint32_t ldsX = lds_off_of(&smX[xr * 64 + xc]);
    const int      tokX = smTok[xr];
    const bf16*    srcXbase = Xb + (size_t)tokX * HDIM + xc;

    for (int kk = 0; kk < HDIM; kk += 64) {
        // X tile: async global -> LDS, no VGPR staging
        async_copy64B(ldsX, srcXbase + kk);
        // W1/W3 tiles: fp32 -> bf16 in flight
        {
            const float* s1 = w1p + (size_t)(n0 + wr) * HDIM + kk + wc;
            const float* s3 = w3p + (size_t)(n0 + wr) * HDIM + kk + wc;
            bf16* d1 = &smW1[wr * 64 + wc];
            bf16* d3 = &smW3[wr * 64 + wc];
#pragma unroll
            for (int q = 0; q < 4; ++q) {
                f32x4 a = *(const f32x4*)(s1 + q * 4);
                f32x4 b = *(const f32x4*)(s3 + q * 4);
#pragma unroll
                for (int i = 0; i < 4; ++i) {
                    d1[q * 4 + i] = (bf16)a[i];
                    d3[q * 4 + i] = (bf16)b[i];
                }
            }
        }
        wait_async0();
        __syncthreads();

#pragma unroll
        for (int kt = 0; kt < 64; kt += 32) {
            v16bf a0  = load_frag64(smX,  wm,      kt, lane);
            v16bf a1  = load_frag64(smX,  wm + 16, kt, lane);
            v16bf b10 = load_frag64(smW1, wn,      kt, lane);
            v16bf b11 = load_frag64(smW1, wn + 16, kt, lane);
            v16bf b30 = load_frag64(smW3, wn,      kt, lane);
            v16bf b31 = load_frag64(smW3, wn + 16, kt, lane);

            accU[0][0] = __builtin_amdgcn_wmma_f32_16x16x32_bf16(false, a0, false, b10, (short)0, accU[0][0], false, false);
            accU[0][1] = __builtin_amdgcn_wmma_f32_16x16x32_bf16(false, a0, false, b11, (short)0, accU[0][1], false, false);
            accU[1][0] = __builtin_amdgcn_wmma_f32_16x16x32_bf16(false, a1, false, b10, (short)0, accU[1][0], false, false);
            accU[1][1] = __builtin_amdgcn_wmma_f32_16x16x32_bf16(false, a1, false, b11, (short)0, accU[1][1], false, false);
            accV[0][0] = __builtin_amdgcn_wmma_f32_16x16x32_bf16(false, a0, false, b30, (short)0, accV[0][0], false, false);
            accV[0][1] = __builtin_amdgcn_wmma_f32_16x16x32_bf16(false, a0, false, b31, (short)0, accV[0][1], false, false);
            accV[1][0] = __builtin_amdgcn_wmma_f32_16x16x32_bf16(false, a1, false, b30, (short)0, accV[1][0], false, false);
            accV[1][1] = __builtin_amdgcn_wmma_f32_16x16x32_bf16(false, a1, false, b31, (short)0, accV[1][1], false, false);
        }
        __syncthreads();
    }

    // epilogue: act = silu(U) * V -> bf16 workspace
    const int rr = (lane >> 4) * 8;              // C/D layout: upper lanes hold M+8
    const int cc = lane & 15;
#pragma unroll
    for (int ri = 0; ri < 2; ++ri)
#pragma unroll
        for (int ci = 0; ci < 2; ++ci) {
            v8f U = accU[ri][ci], V = accV[ri][ci];
#pragma unroll
            for (int g = 0; g < 8; ++g) {
                int m  = wm + ri * 16 + rr + g;
                int lr = rows0 + m;
                if (lr < n) {
                    float a = silu_f(U[g]) * V[g];
                    Act[(size_t)(baseRow + lr) * IDIM + (n0 + wn + ci * 16 + cc)] = (bf16)a;
                }
            }
        }
}

// ---------------- kernel 7: grouped down-projection + gated scatter-add ----------------
__global__ __launch_bounds__(256, 2)
void moe_down_kernel(const bf16* __restrict__ Act, const float* __restrict__ w2,
                     const float* __restrict__ sw2, const int* __restrict__ counts,
                     const int* __restrict__ bases, const int* __restrict__ slotToken,
                     const float* __restrict__ slotGate, float* __restrict__ out) {
    const int v     = blockIdx.z;
    const int n     = (v == NEXP) ? T_TOK : counts[v];
    const int rows0 = blockIdx.x * 128;
    if (rows0 >= n) return;
    const int n0      = blockIdx.y * 64;
    const int baseRow = bases[v];
    const float* w2p = (v == NEXP) ? sw2 : (w2 + (size_t)v * HDIM * IDIM);

    __shared__ __align__(32) bf16 smA[128 * 64];   // 16 KB
    __shared__ __align__(32) bf16 smW[64 * 64];    //  8 KB
    __shared__ int   smTok[128];
    __shared__ float smGate[128];

    const int tid = threadIdx.x;
    if (tid < 128) {
        int lr = rows0 + tid;
        smTok[tid]  = (lr < n) ? slotToken[baseRow + lr] : -1;
        smGate[tid] = (lr < n) ? slotGate[baseRow + lr] : 0.0f;
    }
    __syncthreads();

    const int lane = tid & 31;
    const int wv   = tid >> 5;
    const int wm   = (wv >> 1) * 32;
    const int wn   = (wv & 1) * 32;

    v8f accY[2][2];
#pragma unroll
    for (int i = 0; i < 2; ++i)
#pragma unroll
        for (int j = 0; j < 2; ++j) accY[i][j] = {};

    const int xr = tid >> 1, xc = (tid & 1) * 32;
    const int wr = tid >> 2, wc = (tid & 3) * 16;

    const int lrClamped = min(rows0 + xr, n - 1);     // padded rows read a valid row
    const uint32_t ldsA = lds_off_of(&smA[xr * 64 + xc]);
    const bf16* srcAbase = Act + (size_t)(baseRow + lrClamped) * IDIM + xc;

    for (int kk = 0; kk < IDIM; kk += 64) {
        async_copy64B(ldsA, srcAbase + kk);
        {
            const float* s = w2p + (size_t)(n0 + wr) * IDIM + kk + wc;
            bf16* d = &smW[wr * 64 + wc];
#pragma unroll
            for (int q = 0; q < 4; ++q) {
                f32x4 a = *(const f32x4*)(s + q * 4);
#pragma unroll
                for (int i = 0; i < 4; ++i) d[q * 4 + i] = (bf16)a[i];
            }
        }
        wait_async0();
        __syncthreads();

#pragma unroll
        for (int kt = 0; kt < 64; kt += 32) {
            v16bf a0 = load_frag64(smA, wm,      kt, lane);
            v16bf a1 = load_frag64(smA, wm + 16, kt, lane);
            v16bf b0 = load_frag64(smW, wn,      kt, lane);
            v16bf b1 = load_frag64(smW, wn + 16, kt, lane);

            accY[0][0] = __builtin_amdgcn_wmma_f32_16x16x32_bf16(false, a0, false, b0, (short)0, accY[0][0], false, false);
            accY[0][1] = __builtin_amdgcn_wmma_f32_16x16x32_bf16(false, a0, false, b1, (short)0, accY[0][1], false, false);
            accY[1][0] = __builtin_amdgcn_wmma_f32_16x16x32_bf16(false, a1, false, b0, (short)0, accY[1][0], false, false);
            accY[1][1] = __builtin_amdgcn_wmma_f32_16x16x32_bf16(false, a1, false, b1, (short)0, accY[1][1], false, false);
        }
        __syncthreads();
    }

    const int rr = (lane >> 4) * 8;
    const int cc = lane & 15;
#pragma unroll
    for (int ri = 0; ri < 2; ++ri)
#pragma unroll
        for (int ci = 0; ci < 2; ++ci) {
            v8f Y = accY[ri][ci];
#pragma unroll
            for (int g = 0; g < 8; ++g) {
                int m  = wm + ri * 16 + rr + g;
                int lr = rows0 + m;
                if (lr < n) {
                    int   token = smTok[m];
                    float gte   = smGate[m];
                    atomicAdd(&out[(size_t)token * HDIM + (n0 + wn + ci * 16 + cc)], gte * Y[g]);
                }
            }
        }
}

// ---------------- host launch ----------------
extern "C" void kernel_launch(void* const* d_in, const int* in_sizes, int n_in,
                              void* d_out, int out_size, void* d_ws, size_t ws_size,
                              hipStream_t stream) {
    (void)in_sizes; (void)n_in; (void)out_size; (void)ws_size;
    const float* x    = (const float*)d_in[0];
    const float* gw   = (const float*)d_in[1];
    const float* bias = (const float*)d_in[2];
    const float* sw1  = (const float*)d_in[3];
    const float* sw3  = (const float*)d_in[4];
    const float* sw2  = (const float*)d_in[5];
    const float* w1   = (const float*)d_in[6];
    const float* w3   = (const float*)d_in[7];
    const float* w2   = (const float*)d_in[8];
    float* out = (float*)d_out;

    auto align_up = [](uintptr_t p) { return (p + 255) & ~(uintptr_t)255; };
    uintptr_t p = (uintptr_t)d_ws;
    bf16* Xb        = (bf16*)p;                p = align_up(p + (size_t)T_TOK * HDIM * sizeof(bf16));
    bf16* Act       = (bf16*)p;                p = align_up(p + (size_t)ROWS_TOTAL * IDIM * sizeof(bf16));
    int*   topkIdx  = (int*)p;                 p = align_up(p + (size_t)T_TOK * 2 * sizeof(int));
    float* topkGate = (float*)p;               p = align_up(p + (size_t)T_TOK * 2 * sizeof(float));
    int*   countsP  = (int*)p;                 p = align_up(p + NV * sizeof(int));
    int*   basesP   = (int*)p;                 p = align_up(p + NV * sizeof(int));
    int*   cursorsP = (int*)p;                 p = align_up(p + NV * sizeof(int));
    int*   slotTok  = (int*)p;                 p = align_up(p + (size_t)ROWS_TOTAL * sizeof(int));
    float* slotGate = (float*)p;               p = align_up(p + (size_t)ROWS_TOTAL * sizeof(float));

    moe_zero_kernel<<<2048, 256, 0, stream>>>(out, countsP);
    moe_cvt_kernel<<<4096, 256, 0, stream>>>(x, Xb);
    moe_router_kernel<<<T_TOK / 8, 256, 0, stream>>>(x, gw, bias, topkIdx, topkGate, countsP);
    moe_prefix_kernel<<<1, 32, 0, stream>>>(countsP, basesP, cursorsP);
    moe_scatter_kernel<<<(T_TOK + 255) / 256, 256, 0, stream>>>(topkIdx, topkGate, basesP, cursorsP,
                                                               slotTok, slotGate);
    moe_up_kernel<<<dim3(T_TOK / 128, IDIM / 64, NV), 256, 0, stream>>>(
        Xb, w1, w3, sw1, sw3, countsP, basesP, slotTok, Act);
    moe_down_kernel<<<dim3(T_TOK / 128, HDIM / 64, NV), 256, 0, stream>>>(
        Act, w2, sw2, countsP, basesP, slotTok, slotGate, out);
}
